// MPNEncoder_31791347925653
// MI455X (gfx1250) — compile-verified, matrix operand order checked by reading, not run on hardware
//
#include <hip/hip_runtime.h>
#include <hip/hip_bf16.h>
#include <stdint.h>

// Problem constants (match reference)
#define N_ATOMS_C   60000
#define N_BONDS_C   120000
#define MAX_NB_C    6
#define N_MOLS_C    2000
#define ATOM_FDIM_C 133
#define BOND_FDIM_C 147
#define HIDDEN_C    512
#define KP_BOND_C   160   // 147 padded to multiple of 32
#define CONCAT_C    645   // 133 + 512
#define KP_OUT_C    672   // 645 padded to multiple of 32

typedef __attribute__((ext_vector_type(16))) __bf16 v16bf;
typedef __attribute__((ext_vector_type(8)))  float  v8f;

// ---------- bf16 helpers ----------
// RNE conversion (used once per weight / per stored output element)
__device__ __forceinline__ unsigned short f2bf(float f) {
  union { float f; unsigned u; } x; x.f = f;
  unsigned u = x.u;
  return (unsigned short)((u + 0x7FFFu + ((u >> 16) & 1u)) >> 16);
}
__device__ __forceinline__ float bf2f(unsigned short h) {
  union { unsigned u; float f; } x; x.u = ((unsigned)h) << 16;
  return x.f;
}
// Truncating pack of two f32 into packed bf16x2 (cheap: one v_perm-class op)
__device__ __forceinline__ unsigned pack_bf2_trunc(float lo, float hi) {
  return (__float_as_uint(hi) & 0xFFFF0000u) | (__float_as_uint(lo) >> 16);
}

__device__ __forceinline__ v8f wmma_bf16(v16bf a, v16bf b, v8f c) {
  // D = A(16x32 bf16) * B(32x16 bf16) + C(16x16 f32)
  return __builtin_amdgcn_wmma_f32_16x16x32_bf16(false, a, false, b,
                                                 (short)0, c, false, false);
}

// B-fragment (32x16 bf16, K x N) where B[k][n] = W[n][k] and W is row-major
// [Nout, K] bf16 -> lane l holds column n = ncol + (l&15): contiguous in k.
// Lane K-base = k0 + ((l>>4)<<3); halves[0..7]=K..K+7, halves[8..15]=K+16..K+23.
__device__ __forceinline__ v16bf load_b_frag(const unsigned short* __restrict__ W,
                                             int ldh, int k0, int lane, int ncol) {
  const unsigned short* p = W + (size_t)(ncol + (lane & 15)) * ldh + k0 + ((lane >> 4) << 3);
  union { v16bf v; uint4 q[2]; } u;
  u.q[0] = *(const uint4*)(p);
  u.q[1] = *(const uint4*)(p + 16);
  return u.v;
}

// A-fragment converted on the fly from an f32 row (masked past K).
__device__ __forceinline__ v16bf load_a_f32(const float* __restrict__ row,
                                            int K, int k0, int lane) {
  int kb = k0 + ((lane >> 4) << 3);
  union { v16bf v; unsigned u[8]; } o;
#pragma unroll
  for (int j = 0; j < 4; ++j) {
    int k = kb + 2 * j;
    float lo = (k     < K) ? row[k]     : 0.0f;
    float hi = (k + 1 < K) ? row[k + 1] : 0.0f;
    o.u[j] = pack_bf2_trunc(lo, hi);
    int k2 = kb + 16 + 2 * j;
    lo = (k2     < K) ? row[k2]     : 0.0f;
    hi = (k2 + 1 < K) ? row[k2 + 1] : 0.0f;
    o.u[4 + j] = pack_bf2_trunc(lo, hi);
  }
  return o.v;
}

// Fused directed-bond A-fragment: a_message[b2a[b]] - message[b2revb[b]].
// bf16 in (uint4 loads), subtract in f32 via shifts, truncating repack.
__device__ __forceinline__ v16bf load_a_sub(const unsigned short* __restrict__ pa,
                                            const unsigned short* __restrict__ pr,
                                            int k0, int lane) {
  int kb = k0 + ((lane >> 4) << 3);
  uint4 qa0 = *(const uint4*)(pa + kb);
  uint4 qa1 = *(const uint4*)(pa + kb + 16);
  uint4 qr0 = *(const uint4*)(pr + kb);
  uint4 qr1 = *(const uint4*)(pr + kb + 16);
  const unsigned* a0 = (const unsigned*)&qa0;
  const unsigned* a1 = (const unsigned*)&qa1;
  const unsigned* r0 = (const unsigned*)&qr0;
  const unsigned* r1 = (const unsigned*)&qr1;
  union { v16bf v; unsigned u[8]; } o;
#pragma unroll
  for (int j = 0; j < 4; ++j) {
    float alo = __uint_as_float(a0[j] << 16);
    float ahi = __uint_as_float(a0[j] & 0xFFFF0000u);
    float rlo = __uint_as_float(r0[j] << 16);
    float rhi = __uint_as_float(r0[j] & 0xFFFF0000u);
    o.u[j] = pack_bf2_trunc(alo - rlo, ahi - rhi);
    alo = __uint_as_float(a1[j] << 16);
    ahi = __uint_as_float(a1[j] & 0xFFFF0000u);
    rlo = __uint_as_float(r1[j] << 16);
    rhi = __uint_as_float(r1[j] & 0xFFFF0000u);
    o.u[4 + j] = pack_bf2_trunc(alo - rlo, ahi - rhi);
  }
  return o.v;
}

// Fused concat A-fragment: [f_atoms(133 f32) | a_message(512 bf16) | pad]
__device__ __forceinline__ float concat_elem(const float* __restrict__ fa,
                                             const unsigned short* __restrict__ am,
                                             int k) {
  if (k < ATOM_FDIM_C) return fa[k];
  if (k < CONCAT_C)    return bf2f(am[k - ATOM_FDIM_C]);
  return 0.0f;
}
__device__ __forceinline__ v16bf load_a_concat(const float* __restrict__ fa_row,
                                               const unsigned short* __restrict__ am_row,
                                               int k0, int lane) {
  int kb = k0 + ((lane >> 4) << 3);
  union { v16bf v; unsigned u[8]; } o;
#pragma unroll
  for (int j = 0; j < 4; ++j) {
    int k = kb + 2 * j;
    o.u[j] = pack_bf2_trunc(concat_elem(fa_row, am_row, k),
                            concat_elem(fa_row, am_row, k + 1));
    int k2 = kb + 16 + 2 * j;
    o.u[4 + j] = pack_bf2_trunc(concat_elem(fa_row, am_row, k2),
                                concat_elem(fa_row, am_row, k2 + 1));
  }
  return o.v;
}

// ---------- kernels ----------

// Convert + zero-pad weights to bf16: W_i [512,160], W_h [512,512], W_o [512,672]
__global__ void k_convert_weights(const float* __restrict__ W_i,
                                  const float* __restrict__ W_h,
                                  const float* __restrict__ W_o,
                                  unsigned short* __restrict__ Wi,
                                  unsigned short* __restrict__ Wh,
                                  unsigned short* __restrict__ Wo) {
  const int ni = HIDDEN_C * KP_BOND_C;
  const int nh = HIDDEN_C * HIDDEN_C;
  const int no = HIDDEN_C * KP_OUT_C;
  int i = blockIdx.x * blockDim.x + threadIdx.x;
  if (i < ni) {
    int r = i / KP_BOND_C, c = i % KP_BOND_C;
    Wi[i] = (c < BOND_FDIM_C) ? f2bf(W_i[(size_t)r * BOND_FDIM_C + c]) : (unsigned short)0;
  } else if (i < ni + nh) {
    int j = i - ni;
    Wh[j] = f2bf(W_h[j]);
  } else if (i < ni + nh + no) {
    int j = i - ni - nh;
    int r = j / KP_OUT_C, c = j % KP_OUT_C;
    Wo[j] = (c < CONCAT_C) ? f2bf(W_o[(size_t)r * CONCAT_C + c]) : (unsigned short)0;
  }
}

__global__ void k_zero_f32(float* __restrict__ p, int n) {
  int i = blockIdx.x * blockDim.x + threadIdx.x;
  if (i < n) p[i] = 0.0f;
}

// inp = f_bonds @ W_i.T ; message = relu(inp).
// Block = 256 thr = 8 waves; block tile = 32 rows x 512 cols.
// Waves 0-3 -> M-tile 0 cols {0,128,256,384}; waves 4-7 -> M-tile 1.
// Each wave: 16x128 output (8 WMMA tiles), A-fragment reused 8x.
__global__ void __launch_bounds__(256)
k_gemm_input(const float* __restrict__ fb, const unsigned short* __restrict__ Wi,
             unsigned short* __restrict__ inp, unsigned short* __restrict__ msg) {
  int lane = threadIdx.x & 31;
  int wave = threadIdx.x >> 5;
  int m0 = blockIdx.x * 32 + (wave >> 2) * 16;
  int n0 = (wave & 3) * 128;
  const float* arow = fb + (size_t)(m0 + (lane & 15)) * BOND_FDIM_C;
  v8f acc[8] = {};
  for (int k0 = 0; k0 < BOND_FDIM_C; k0 += 32) {
    v16bf a = load_a_f32(arow, BOND_FDIM_C, k0, lane);
#pragma unroll
    for (int t = 0; t < 8; ++t) {
      v16bf b = load_b_frag(Wi, KP_BOND_C, k0, lane, n0 + t * 16);
      acc[t] = wmma_bf16(a, b, acc[t]);
    }
  }
  int rb = (lane >> 4) << 3;
#pragma unroll
  for (int t = 0; t < 8; ++t) {
    int col = n0 + t * 16 + (lane & 15);
#pragma unroll
    for (int i = 0; i < 8; ++i) {
      size_t o = (size_t)(m0 + rb + i) * HIDDEN_C + col;
      float v = acc[t][i];
      inp[o] = f2bf(v);
      msg[o] = f2bf(fmaxf(v, 0.0f));
    }
  }
}

// a_message[atom] = sum_{j<6} message[a2b[atom][j]]  (bf16 in, f32 acc, bf16 out)
__global__ void k_gather_sum(const unsigned short* __restrict__ msg,
                             const int* __restrict__ a2b,
                             unsigned short* __restrict__ amsg) {
  int tid = blockIdx.x * blockDim.x + threadIdx.x;
  if (tid >= N_ATOMS_C * (HIDDEN_C / 8)) return;
  int atom = tid >> 6;
  int c8 = (tid & 63) << 3;
  float acc[8] = {};
#pragma unroll
  for (int j = 0; j < MAX_NB_C; ++j) {
    int b = a2b[atom * MAX_NB_C + j];
    uint4 q = *(const uint4*)(msg + (size_t)b * HIDDEN_C + c8);
    const unsigned short* h = (const unsigned short*)&q;
#pragma unroll
    for (int e = 0; e < 8; ++e) acc[e] += bf2f(h[e]);
  }
  union { uint4 q; unsigned short s[8]; } o;
#pragma unroll
  for (int e = 0; e < 8; ++e) o.s[e] = f2bf(acc[e]);
  *(uint4*)(amsg + (size_t)atom * HIDDEN_C + c8) = o.q;
}

// message_out = relu(inp + (a_message[b2a] - message[b2revb]) @ W_h.T)
// gather/subtract fused into the A-fragment load; A reused across 8 N-tiles.
__global__ void __launch_bounds__(256)
k_gemm_bond(const unsigned short* __restrict__ amsg,
            const unsigned short* __restrict__ msg_in,
            const unsigned short* __restrict__ inp,
            const int* __restrict__ b2a, const int* __restrict__ b2revb,
            const unsigned short* __restrict__ Wh,
            unsigned short* __restrict__ msg_out) {
  int lane = threadIdx.x & 31;
  int wave = threadIdx.x >> 5;
  int m0 = blockIdx.x * 32 + (wave >> 2) * 16;
  int n0 = (wave & 3) * 128;
  int row = m0 + (lane & 15);
  const unsigned short* pa = amsg   + (size_t)b2a[row]    * HIDDEN_C;
  const unsigned short* pr = msg_in + (size_t)b2revb[row] * HIDDEN_C;
  v8f acc[8] = {};
  for (int k0 = 0; k0 < HIDDEN_C; k0 += 32) {
    __builtin_prefetch((const void*)(pa + k0 + 32), 0, 1);
    __builtin_prefetch((const void*)(pr + k0 + 32), 0, 1);
    v16bf a = load_a_sub(pa, pr, k0, lane);
#pragma unroll
    for (int t = 0; t < 8; ++t) {
      v16bf b = load_b_frag(Wh, HIDDEN_C, k0, lane, n0 + t * 16);
      acc[t] = wmma_bf16(a, b, acc[t]);
    }
  }
  int rb = (lane >> 4) << 3;
#pragma unroll
  for (int t = 0; t < 8; ++t) {
    int col = n0 + t * 16 + (lane & 15);
#pragma unroll
    for (int i = 0; i < 8; ++i) {
      size_t o = (size_t)(m0 + rb + i) * HIDDEN_C + col;
      float v = bf2f(inp[o]) + acc[t][i];
      msg_out[o] = f2bf(fmaxf(v, 0.0f));
    }
  }
}

// atom_hiddens = relu([f_atoms | a_message] @ W_o.T + b_o); per-molecule
// max-pool via int-bit atomicMax (valid: values >= 0, d_out pre-zeroed).
__global__ void __launch_bounds__(256)
k_gemm_out_pool(const float* __restrict__ f_atoms,
                const unsigned short* __restrict__ amsg,
                const unsigned short* __restrict__ Wo,
                const float* __restrict__ b_o,
                const int* __restrict__ atom2mol,
                float* __restrict__ out) {
  int lane = threadIdx.x & 31;
  int wave = threadIdx.x >> 5;
  int m0 = blockIdx.x * 32 + (wave >> 2) * 16;
  int n0 = (wave & 3) * 128;
  int row = m0 + (lane & 15);
  const float* farow = f_atoms + (size_t)row * ATOM_FDIM_C;
  const unsigned short* amrow = amsg + (size_t)row * HIDDEN_C;
  v8f acc[8] = {};
  for (int k0 = 0; k0 < KP_OUT_C; k0 += 32) {
    v16bf a = load_a_concat(farow, amrow, k0, lane);
#pragma unroll
    for (int t = 0; t < 8; ++t) {
      v16bf b = load_b_frag(Wo, KP_OUT_C, k0, lane, n0 + t * 16);
      acc[t] = wmma_bf16(a, b, acc[t]);
    }
  }
  int rb = (lane >> 4) << 3;
#pragma unroll
  for (int t = 0; t < 8; ++t) {
    int col = n0 + t * 16 + (lane & 15);
    float bias = b_o[col];
#pragma unroll
    for (int i = 0; i < 8; ++i) {
      int r = m0 + rb + i;
      float v = fmaxf(acc[t][i] + bias, 0.0f);
      int mol = atom2mol[r];
      atomicMax((int*)&out[(size_t)mol * HIDDEN_C + col], __float_as_int(v));
    }
  }
}

// ---------- host launch ----------
extern "C" void kernel_launch(void* const* d_in, const int* in_sizes, int n_in,
                              void* d_out, int out_size, void* d_ws, size_t ws_size,
                              hipStream_t stream) {
  const float* f_atoms = (const float*)d_in[0];
  const float* f_bonds = (const float*)d_in[1];
  const int*   a2b     = (const int*)d_in[2];
  const int*   b2a     = (const int*)d_in[3];
  const int*   b2revb  = (const int*)d_in[4];
  const int*   a2mol   = (const int*)d_in[5];
  const float* W_i     = (const float*)d_in[6];
  const float* W_h     = (const float*)d_in[7];
  const float* W_o     = (const float*)d_in[8];
  const float* b_o     = (const float*)d_in[9];
  float* out = (float*)d_out;

  // workspace carve (all 256B aligned)
  char* ws = (char*)d_ws;
  size_t off = 0;
  auto carve = [&](size_t bytes) { void* p = ws + off; off += (bytes + 255) & ~(size_t)255; return p; };
  unsigned short* Wi   = (unsigned short*)carve((size_t)HIDDEN_C * KP_BOND_C * 2);
  unsigned short* Wh   = (unsigned short*)carve((size_t)HIDDEN_C * HIDDEN_C  * 2);
  unsigned short* Wo   = (unsigned short*)carve((size_t)HIDDEN_C * KP_OUT_C  * 2);
  unsigned short* inp  = (unsigned short*)carve((size_t)N_BONDS_C * HIDDEN_C * 2);
  unsigned short* msgA = (unsigned short*)carve((size_t)N_BONDS_C * HIDDEN_C * 2);
  unsigned short* msgB = (unsigned short*)carve((size_t)N_BONDS_C * HIDDEN_C * 2);
  unsigned short* amsg = (unsigned short*)carve((size_t)N_ATOMS_C * HIDDEN_C * 2);
  (void)ws_size; (void)n_in; (void)in_sizes;

  const int wtotal = HIDDEN_C * (KP_BOND_C + HIDDEN_C + KP_OUT_C);
  k_convert_weights<<<(wtotal + 255) / 256, 256, 0, stream>>>(W_i, W_h, W_o, Wi, Wh, Wo);
  k_zero_f32<<<(out_size + 255) / 256, 256, 0, stream>>>(out, out_size);

  // message = relu(inp = f_bonds @ W_i.T)
  k_gemm_input<<<N_BONDS_C / 32, 256, 0, stream>>>(f_bonds, Wi, inp, msgA);

  const int gather_blocks = (N_ATOMS_C * (HIDDEN_C / 8) + 255) / 256;
  // depth iteration 1
  k_gather_sum<<<gather_blocks, 256, 0, stream>>>(msgA, a2b, amsg);
  k_gemm_bond<<<N_BONDS_C / 32, 256, 0, stream>>>(amsg, msgA, inp, b2a, b2revb, Wh, msgB);
  // depth iteration 2
  k_gather_sum<<<gather_blocks, 256, 0, stream>>>(msgB, a2b, amsg);
  k_gemm_bond<<<N_BONDS_C / 32, 256, 0, stream>>>(amsg, msgB, inp, b2a, b2revb, Wh, msgA);
  // final aggregation
  k_gather_sum<<<gather_blocks, 256, 0, stream>>>(msgA, a2b, amsg);

  // output GEMM + bias + relu + segment max-pool
  k_gemm_out_pool<<<N_ATOMS_C / 32, 256, 0, stream>>>(f_atoms, amsg, Wo, b_o, a2mol, out);
}
